// GeometryRelationAwareModule_55465207661221
// MI455X (gfx1250) — matrix operand order repaired and checked by reading, not compiled, not run on hardware
//
#include <hip/hip_runtime.h>
#include <hip/hip_bf16.h>

// ---------------------------------------------------------------------------
// Problem constants (from the reference)
// ---------------------------------------------------------------------------
#define BB       4
#define NN       512
#define MM       (BB * NN)          // 2048
#define K_FEAT   12544              // 256*7*7, 392 chunks of 32
#define N_OUT    256
#define EMBD     64
#define EPS_F    0.001f
#define LDSW     40                 // LDS row stride in ushorts (80B, 16B-aligned)

typedef __attribute__((ext_vector_type(16))) __bf16 v16bf;
typedef __attribute__((ext_vector_type(8)))  float  v8f;

union FragBF {
    v16bf v;
    unsigned long long q[4];
};

// fp32 -> bf16 (round-to-nearest, ties-away): 1 VALU per element
static __device__ __forceinline__ unsigned short f2bf(float x) {
    return (unsigned short)((__float_as_uint(x) + 0x8000u) >> 16);
}

// two fp32 -> packed bf16x2 in one v_perm_b32: {hi16(hi), hi16(lo)}
static __device__ __forceinline__ unsigned cvt2bf(float lo, float hi) {
    const unsigned a = __float_as_uint(lo) + 0x8000u;
    const unsigned b = __float_as_uint(hi) + 0x8000u;
    return __builtin_amdgcn_perm(b, a, 0x07060302u);
}

static __device__ __forceinline__ unsigned long long pack4bf(float4 f) {
    return (unsigned long long)cvt2bf(f.x, f.y)
         | ((unsigned long long)cvt2bf(f.z, f.w) << 32);
}

// --- CDNA5 async copy to LDS (ASYNCcnt-tracked), per ISA 15.18.3 op 98 ------
#define ASYNC_LOAD_B128(ldsOff, gAddr)                                        \
    asm volatile("global_load_async_to_lds_b128 %0, %1, off"                  \
                 :: "v"(ldsOff), "v"(gAddr) : "memory")

#if __has_builtin(__builtin_amdgcn_s_wait_asynccnt)
#define WAIT_ASYNC(n) __builtin_amdgcn_s_wait_asynccnt(n)
#else
#define WAIT_ASYNC(n) asm volatile("s_wait_asynccnt %0" :: "i"(n) : "memory")
#endif

// ---------------------------------------------------------------------------
// Kernel 1: pairwise geometry -> sin/cos embedding -> score -> softmax row.
// One block per (b, i); writes attn row in bf16.
// ---------------------------------------------------------------------------
__global__ void attn_row_kernel(const float* __restrict__ rois,
                                const float* __restrict__ W1,
                                const float* __restrict__ b1,
                                unsigned short* __restrict__ attnb) {
    __shared__ float scx[NN], scy[NN], sw[NN], sh[NN];
    __shared__ float sW1[EMBD];
    __shared__ float srow[NN];
    __shared__ float red[256];

    const int i   = blockIdx.x;
    const int b   = blockIdx.y;
    const int tid = threadIdx.x;

    for (int j = tid; j < NN; j += 256) {
        const float4 r = *(const float4*)(rois + ((size_t)(b * NN + j)) * 4);
        scx[j] = (r.x + r.z) * 0.5f;
        scy[j] = (r.y + r.w) * 0.5f;
        sw[j]  = r.z - r.x;
        sh[j]  = r.w - r.y;
    }
    if (tid < EMBD) sW1[tid] = W1[tid];
    __syncthreads();

    const float cxi = scx[i], cyi = scy[i], wi = sw[i], hi = sh[i];
    const float b1v = b1[0];

    const float invf[8] = {
        1.0f, 1.0f / 2.37137371f, 1.0f / 5.62341325f, 1.0f / 13.3352143f,
        1.0f / 31.6227766f, 1.0f / 74.9894209f, 1.0f / 177.827941f, 1.0f / 421.696503f
    };

    for (int j = tid; j < NN; j += 256) {
        const float gw = fmaxf(sw[j], EPS_F);
        const float gh = fmaxf(sh[j], EPS_F);
        const float rx = fmaxf(fabsf(scx[j] - cxi) / gw, EPS_F);
        const float ry = fmaxf(fabsf(scy[j] - cyi) / gh, EPS_F);
        const float rw = fmaxf(wi / gw, EPS_F);
        const float rh = fmaxf(hi / gh, EPS_F);
        float geo[4];
        geo[0] = 100.0f * __logf(rx);
        geo[1] = 100.0f * __logf(ry);
        geo[2] = 100.0f * __logf(rw);
        geo[3] = 100.0f * __logf(rh);
        float s = b1v;
#pragma unroll
        for (int c = 0; c < 4; ++c)
#pragma unroll
            for (int f = 0; f < 8; ++f) {
                const float arg = geo[c] * invf[f];
                float sv, cv;
                __sincosf(arg, &sv, &cv);
                s += sv * sW1[c * 16 + 2 * f] + cv * sW1[c * 16 + 2 * f + 1];
            }
        srow[j] = fmaxf(s, 0.0f);
    }
    __syncthreads();

    float m = -1e30f;
    for (int j = tid; j < NN; j += 256) m = fmaxf(m, srow[j]);
    red[tid] = m;
    __syncthreads();
#pragma unroll
    for (int st = 128; st > 0; st >>= 1) {
        if (tid < st) red[tid] = fmaxf(red[tid], red[tid + st]);
        __syncthreads();
    }
    const float mx = red[0];
    __syncthreads();

    float ssum = 0.0f;
    for (int j = tid; j < NN; j += 256) {
        const float e = __expf(srow[j] - mx);
        srow[j] = e;
        ssum += e;
    }
    red[tid] = ssum;
    __syncthreads();
#pragma unroll
    for (int st = 128; st > 0; st >>= 1) {
        if (tid < st) red[tid] += red[tid + st];
        __syncthreads();
    }
    const float inv = 1.0f / red[0];

    unsigned short* out = attnb + ((size_t)(b * NN + i)) * NN;
    for (int j = tid; j < NN; j += 256) out[j] = f2bf(srow[j] * inv);
}

// ---------------------------------------------------------------------------
// Kernel 2: featsT = (gt_feats @ W2^T + b2)^T  in bf16, layout [b][n=256][j=512]
// M=2048, N=256, K=12544. Register-pipelined: load tile k+1 while WMMAs run on
// tile k. Block tile 128x64, 8 waves, 2x2 v_wmma_f32_16x16x32_bf16 per wave.
// ---------------------------------------------------------------------------
__global__ void feats_gemm_kernel(const float* __restrict__ A,
                                  const float* __restrict__ Bm,   // W2 [256][12544]
                                  const float* __restrict__ bias,
                                  unsigned short* __restrict__ featsT) {
    __shared__ unsigned short As[128 * LDSW];
    __shared__ unsigned short Bs[64 * LDSW];

    const int tileM = blockIdx.x * 128;
    const int tileN = blockIdx.y * 64;
    const int tid   = threadIdx.x;
    const int lane  = tid & 31;
    const int wave  = tid >> 5;
    const int wm    = wave >> 1;
    const int wn    = wave & 1;

    v8f acc[2][2] = {};

    // fragment addressing (ISA 7.12.2)
    const int mA = lane & 15;
    const int kh = (lane >> 4) * 8;
    const int nB = lane & 15;
    const int kb = (lane >> 4) * 8;

    // staging maps
    const int aRow = tid >> 3;            // +32 per it
    const int aC4  = (tid & 7) * 4;
    const int bN   = tid >> 3;
    const int bC4  = aC4;

    float4 aReg[4], bReg[2];
#pragma unroll
    for (int it = 0; it < 4; ++it)
        aReg[it] = *(const float4*)(A + (long)(tileM + aRow + it * 32) * K_FEAT + aC4);
#pragma unroll
    for (int it = 0; it < 2; ++it)
        bReg[it] = *(const float4*)(Bm + (long)(tileN + bN + it * 32) * K_FEAT + bC4);

    for (int k0 = 0; k0 < K_FEAT; k0 += 32) {
        // ---- commit staged registers to LDS (v_perm-packed b64 stores)
#pragma unroll
        for (int it = 0; it < 4; ++it)
            *(unsigned long long*)&As[(aRow + it * 32) * LDSW + aC4] = pack4bf(aReg[it]);
#pragma unroll
        for (int it = 0; it < 2; ++it)
            *(unsigned long long*)&Bs[(bN + it * 32) * LDSW + bC4] = pack4bf(bReg[it]);
        __syncthreads();

        // ---- start next tile's global loads (overlap with WMMA below)
        if (k0 + 32 < K_FEAT) {
            const int kn = k0 + 32;
#pragma unroll
            for (int it = 0; it < 4; ++it)
                aReg[it] = *(const float4*)(A + (long)(tileM + aRow + it * 32) * K_FEAT + kn + aC4);
#pragma unroll
            for (int it = 0; it < 2; ++it)
                bReg[it] = *(const float4*)(Bm + (long)(tileN + bN + it * 32) * K_FEAT + kn + bC4);
        }
        // ---- L2 prefetch two tiles ahead (global_prefetch_b8)
        if (k0 + 64 < K_FEAT) {
            __builtin_prefetch(A + (long)(tileM + (tid >> 1)) * K_FEAT + (k0 + 64) + (tid & 1) * 16, 0, 1);
            __builtin_prefetch(Bm + (long)(tileN + (tid >> 2)) * K_FEAT + (k0 + 64) + (tid & 3) * 8, 0, 1);
        }

        // ---- fragments + WMMA
        FragBF afr[2], bfr[2];
#pragma unroll
        for (int ii = 0; ii < 2; ++ii) {
            const unsigned short* ap = &As[(wm * 32 + ii * 16 + mA) * LDSW];
            afr[ii].q[0] = *(const unsigned long long*)(ap + kh);
            afr[ii].q[1] = *(const unsigned long long*)(ap + kh + 4);
            afr[ii].q[2] = *(const unsigned long long*)(ap + 16 + kh);
            afr[ii].q[3] = *(const unsigned long long*)(ap + 16 + kh + 4);
        }
#pragma unroll
        for (int jj = 0; jj < 2; ++jj) {
            const unsigned short* bp = &Bs[(wn * 32 + jj * 16 + nB) * LDSW + kb];
            bfr[jj].q[0] = *(const unsigned long long*)(bp);
            bfr[jj].q[1] = *(const unsigned long long*)(bp + 4);
            bfr[jj].q[2] = *(const unsigned long long*)(bp + 16);
            bfr[jj].q[3] = *(const unsigned long long*)(bp + 20);
        }
#pragma unroll
        for (int ii = 0; ii < 2; ++ii)
#pragma unroll
            for (int jj = 0; jj < 2; ++jj)
                acc[ii][jj] = __builtin_amdgcn_wmma_f32_16x16x32_bf16(
                    false, afr[ii].v, false, bfr[jj].v,
                    (short)0, acc[ii][jj], false, false);
        __syncthreads();
    }

    // ---- epilogue: bias add, bf16 via v_perm, store transposed [b][n][j]
    const int nC    = lane & 15;
    const int mHalf = (lane >> 4) * 8;
#pragma unroll
    for (int ii = 0; ii < 2; ++ii)
#pragma unroll
        for (int jj = 0; jj < 2; ++jj) {
            const int gm0 = tileM + wm * 32 + ii * 16 + mHalf;   // multiple of 8
            const int gn  = tileN + wn * 32 + jj * 16 + nC;
            const int b   = gm0 >> 9;
            const int j0  = gm0 & 511;
            const float bv = bias[gn];
            uint4 pk;
            pk.x = cvt2bf(acc[ii][jj][0] + bv, acc[ii][jj][1] + bv);
            pk.y = cvt2bf(acc[ii][jj][2] + bv, acc[ii][jj][3] + bv);
            pk.z = cvt2bf(acc[ii][jj][4] + bv, acc[ii][jj][5] + bv);
            pk.w = cvt2bf(acc[ii][jj][6] + bv, acc[ii][jj][7] + bv);
            *(uint4*)(featsT + ((size_t)(b * N_OUT + gn) * NN + j0)) = pk;
        }
}

// ---------------------------------------------------------------------------
// Kernel 3: out[b] = attn[b] @ feats[b].  Both operands already bf16 in the
// exact [row][k] layouts needed, so A and B tiles are staged with
// global_load_async_to_lds_b128 (ASYNCcnt), double-buffered in LDS.
// Per wave per tile: 3 async ops -> s_wait_asynccnt 3 retires previous tile.
// ---------------------------------------------------------------------------
__global__ void out_gemm_async_kernel(const unsigned short* __restrict__ attnb,   // [b][i][j]
                                      const unsigned short* __restrict__ featsT,  // [b][n][j]
                                      float* __restrict__ C) {                    // [b*512][256]
    __shared__ unsigned short AsBuf[2][128 * LDSW];
    __shared__ unsigned short BsBuf[2][64 * LDSW];

    const int b     = blockIdx.z;
    const int tileM = blockIdx.x * 128;
    const int tileN = blockIdx.y * 64;
    const int tid   = threadIdx.x;
    const int lane  = tid & 31;
    const int wave  = tid >> 5;
    const int wm    = wave >> 1;
    const int wn    = wave & 1;

    v8f acc[2][2] = {};

    const int mA = lane & 15;
    const int kh = (lane >> 4) * 8;
    const int nB = lane & 15;
    const int kb = (lane >> 4) * 8;

    // async staging maps: 16B granules (8 bf16 along k)
    const int am  = tid >> 2;        // 0..63 ; +64 for second granule
    const int seg = (tid & 3) * 8;   // k offset in elements
    const int bn  = tid >> 2;        // 0..63

    const unsigned aLds[2] = { (unsigned)(uintptr_t)&AsBuf[0][0], (unsigned)(uintptr_t)&AsBuf[1][0] };
    const unsigned bLds[2] = { (unsigned)(uintptr_t)&BsBuf[0][0], (unsigned)(uintptr_t)&BsBuf[1][0] };

    const unsigned long long aG =
        (unsigned long long)(uintptr_t)(attnb) + 2ull * ((unsigned long long)(b * NN + tileM) * NN);
    const unsigned long long bG =
        (unsigned long long)(uintptr_t)(featsT) + 2ull * ((unsigned long long)(b * N_OUT + tileN) * NN);

    const int NT = NN / 32;   // 16 K-tiles

    // prologue: issue tile 0 into buffer 0
    ASYNC_LOAD_B128(aLds[0] + (unsigned)(am * 2 * LDSW + seg * 2),
                    aG + 2ull * ((unsigned long long)am * NN + seg));
    ASYNC_LOAD_B128(aLds[0] + (unsigned)((am + 64) * 2 * LDSW + seg * 2),
                    aG + 2ull * ((unsigned long long)(am + 64) * NN + seg));
    ASYNC_LOAD_B128(bLds[0] + (unsigned)(bn * 2 * LDSW + seg * 2),
                    bG + 2ull * ((unsigned long long)bn * NN + seg));

    for (int kt = 0; kt < NT; ++kt) {
        const int cur = kt & 1;
        if (kt + 1 < NT) {
            const int nxt = cur ^ 1;
            const int k0  = (kt + 1) * 32;
            ASYNC_LOAD_B128(aLds[nxt] + (unsigned)(am * 2 * LDSW + seg * 2),
                            aG + 2ull * ((unsigned long long)am * NN + k0 + seg));
            ASYNC_LOAD_B128(aLds[nxt] + (unsigned)((am + 64) * 2 * LDSW + seg * 2),
                            aG + 2ull * ((unsigned long long)(am + 64) * NN + k0 + seg));
            ASYNC_LOAD_B128(bLds[nxt] + (unsigned)(bn * 2 * LDSW + seg * 2),
                            bG + 2ull * ((unsigned long long)bn * NN + k0 + seg));
            WAIT_ASYNC(3);   // previous tile's 3 ops retired (in-order completion)
        } else {
            WAIT_ASYNC(0);
        }
        __syncthreads();

        const unsigned short* AsP = &AsBuf[cur][0];
        const unsigned short* BsP = &BsBuf[cur][0];

        FragBF afr[2], bfr[2];
#pragma unroll
        for (int ii = 0; ii < 2; ++ii) {
            const unsigned short* ap = AsP + (wm * 32 + ii * 16 + mA) * LDSW;
            afr[ii].q[0] = *(const unsigned long long*)(ap + kh);
            afr[ii].q[1] = *(const unsigned long long*)(ap + kh + 4);
            afr[ii].q[2] = *(const unsigned long long*)(ap + 16 + kh);
            afr[ii].q[3] = *(const unsigned long long*)(ap + 16 + kh + 4);
        }
#pragma unroll
        for (int jj = 0; jj < 2; ++jj) {
            const unsigned short* bp = BsP + (wn * 32 + jj * 16 + nB) * LDSW + kb;
            bfr[jj].q[0] = *(const unsigned long long*)(bp);
            bfr[jj].q[1] = *(const unsigned long long*)(bp + 4);
            bfr[jj].q[2] = *(const unsigned long long*)(bp + 16);
            bfr[jj].q[3] = *(const unsigned long long*)(bp + 20);
        }
#pragma unroll
        for (int ii = 0; ii < 2; ++ii)
#pragma unroll
            for (int jj = 0; jj < 2; ++jj)
                acc[ii][jj] = __builtin_amdgcn_wmma_f32_16x16x32_bf16(
                    false, afr[ii].v, false, bfr[jj].v,
                    (short)0, acc[ii][jj], false, false);
        __syncthreads();
    }

    const int nC    = lane & 15;
    const int mHalf = (lane >> 4) * 8;
#pragma unroll
    for (int ii = 0; ii < 2; ++ii)
#pragma unroll
        for (int jj = 0; jj < 2; ++jj) {
            const int gm0 = tileM + wm * 32 + ii * 16 + mHalf;
            const int gn  = tileN + wn * 32 + jj * 16 + nC;
#pragma unroll
            for (int r = 0; r < 8; ++r)
                C[(size_t)(b * NN + gm0 + r) * N_OUT + gn] = acc[ii][jj][r];
        }
}

// ---------------------------------------------------------------------------
// Launch
// ---------------------------------------------------------------------------
extern "C" void kernel_launch(void* const* d_in, const int* in_sizes, int n_in,
                              void* d_out, int out_size, void* d_ws, size_t ws_size,
                              hipStream_t stream) {
    const float* rois     = (const float*)d_in[0];
    // d_in[1] = gt_rois : unused (reference overwrites it with rois)
    const float* gt_feats = (const float*)d_in[2];   // [2048, 12544]
    const float* W1       = (const float*)d_in[3];   // [1, 64]
    const float* b1       = (const float*)d_in[4];   // [1]
    const float* W2       = (const float*)d_in[5];   // [256, 12544]
    const float* b2       = (const float*)d_in[6];   // [256]

    unsigned short* attnb  = (unsigned short*)d_ws;                       // [B,N,N] bf16 = 2 MB
    unsigned short* featsT = attnb + (size_t)MM * NN;                     // [B,256,N] bf16 = 1 MB
    float* out = (float*)d_out;

    attn_row_kernel<<<dim3(NN, BB), 256, 0, stream>>>(rois, W1, b1, attnb);

    feats_gemm_kernel<<<dim3(MM / 128, N_OUT / 64, 1), 256, 0, stream>>>(
        gt_feats, W2, b2, featsT);

    out_gemm_async_kernel<<<dim3(NN / 128, N_OUT / 64, BB), 256, 0, stream>>>(
        attnb, featsT, out);
}
